// GNNEncoder_11957188952733
// MI455X (gfx1250) — matrix operand order repaired and checked by reading, not compiled, not run on hardware
//
#include <hip/hip_runtime.h>
#include <hip/hip_bf16.h>

typedef __attribute__((ext_vector_type(16))) __bf16 v16bf;
typedef __attribute__((ext_vector_type(8)))  float  v8f;

#define IN_C 128
#define HID_C 128

// ---------------------------------------------------------------------------
// zero fill
// ---------------------------------------------------------------------------
__global__ void sage_zero_f32(float* __restrict__ p, long long n) {
    long long i = (long long)blockIdx.x * blockDim.x + threadIdx.x;
    long long stride = (long long)gridDim.x * blockDim.x;
    for (; i < n; i += stride) p[i] = 0.0f;
}

// ---------------------------------------------------------------------------
// one-time fp32 -> bf16 weight conversion into workspace
// ---------------------------------------------------------------------------
__global__ void sage_w2bf(const float* __restrict__ w, __bf16* __restrict__ o, int n) {
    int i = blockIdx.x * blockDim.x + threadIdx.x;
    if (i < n) o[i] = (__bf16)w[i];
}

// ---------------------------------------------------------------------------
// scatter-add of source features into sum[dst], plus degree count.
// one thread per (edge, float4 channel group); C == 128 -> 32 groups.
// ---------------------------------------------------------------------------
__global__ void sage_scatter(const float* __restrict__ feat,
                             const long long* __restrict__ ei,
                             float* __restrict__ sum,
                             float* __restrict__ deg,
                             long long E) {
    long long idx = (long long)blockIdx.x * blockDim.x + threadIdx.x;
    long long total = E * 32;
    if (idx >= total) return;
    long long e = idx >> 5;
    int g = (int)(idx & 31);
    long long s = ei[e];       // src row of edge_index
    long long d = ei[E + e];   // dst row
    float4 v = reinterpret_cast<const float4*>(feat + s * IN_C)[g];
    float* o = sum + d * IN_C + (long long)g * 4;
    atomicAdd(o + 0, v.x);
    atomicAdd(o + 1, v.y);
    atomicAdd(o + 2, v.z);
    atomicAdd(o + 3, v.w);
    if (g == 0) atomicAdd(deg + d, 1.0f);
}

// ---------------------------------------------------------------------------
// WMMA fragment helpers (wave32, v_wmma_f32_16x16x32_bf16)
//
// A (16x32 bf16, MxK): lane L<16 holds row M=L with K in {0..7, 16..23};
//                      lane L>=16 holds row M=L-16 with K in {8..15, 24..31}.
// element i -> K = (i&7) + ((i>>3)<<4) + kh*8   (kh = lane>>4)
// Raw loads are 4x float4 at K byte offsets {0,16,64,80} (+kh*32).
// ---------------------------------------------------------------------------
struct AF32 { float4 q[4]; };

__device__ inline AF32 load_a_raw(const float* __restrict__ p) {
    AF32 r;
    r.q[0] = *reinterpret_cast<const float4*>(p + 0);   // K 0..3
    r.q[1] = *reinterpret_cast<const float4*>(p + 4);   // K 4..7
    r.q[2] = *reinterpret_cast<const float4*>(p + 16);  // K 16..19
    r.q[3] = *reinterpret_cast<const float4*>(p + 20);  // K 20..23
    return r;
}

__device__ inline v16bf cvt_a_frag(const AF32& r, float s) {
    v16bf a;
#pragma unroll
    for (int j = 0; j < 4; ++j) {
        a[4 * j + 0] = (__bf16)(r.q[j].x * s);
        a[4 * j + 1] = (__bf16)(r.q[j].y * s);
        a[4 * j + 2] = (__bf16)(r.q[j].z * s);
        a[4 * j + 3] = (__bf16)(r.q[j].w * s);
    }
    return a;
}

// B (32x16 bf16, KxN): lane L holds column N=L&15; element i -> K = (i&7) + ((i>>3)<<4) + kh*8.
__device__ inline v16bf load_b_frag(const __bf16* __restrict__ p, int ld) {
    v16bf b;
#pragma unroll
    for (int i = 0; i < 16; ++i) b[i] = p[((i & 7) + ((i >> 3) << 4)) * ld];
    return b;
}

// async-copy `bytes` from global bf16 weights into LDS (ASYNCcnt path).
// LDS generic pointer truncated to 32 bits == LDS byte address (ISA aperture
// rule). GVS addressing: mem = SADDR(sgpr pair) + VADDR(u32) ; 16B per lane.
__device__ inline void async_stage(const __bf16* __restrict__ gsrc,
                                   __bf16* lds_dst, int bytes, int tid) {
    unsigned ldsbase = (unsigned)(unsigned long long)lds_dst;
    for (int t = tid; t < bytes / 16; t += 256) {
        unsigned off = (unsigned)t * 16u;
        unsigned lds_addr = ldsbase + off;
        asm volatile("global_load_async_to_lds_b128 %0, %1, %2"
                     :: "v"(lds_addr), "v"(off), "s"(gsrc) : "memory");
    }
}

// ---------------------------------------------------------------------------
// Fused SAGEConv GEMM:  out = act( mean @ Wl + x @ Wr + b )
//   mean formed on the fly from (sum, deg).  K depth = 128.
//   256 threads = 8 waves; block covers 128 node rows; wave w covers 16 rows.
// ---------------------------------------------------------------------------
template <int OUTC, bool RELU>
__global__ __launch_bounds__(256) void sage_gemm(
    const float* __restrict__ sum, const float* __restrict__ deg,
    const float* __restrict__ xin, const __bf16* __restrict__ Wl,
    const __bf16* __restrict__ Wr, const float* __restrict__ bias,
    float* __restrict__ out, long long n_nodes) {
    constexpr int KD = 128;
    constexpr int NT = OUTC / 16;

    __shared__ __bf16 sWl[KD * OUTC];
    __shared__ __bf16 sWr[KD * OUTC];

    const int tid = threadIdx.x;

    // stage pre-converted bf16 weights into LDS via async DMA path
    async_stage(Wl, sWl, KD * OUTC * 2, tid);
    async_stage(Wr, sWr, KD * OUTC * 2, tid);

    const int lane = tid & 31;
    const int wv   = tid >> 5;
    const int m16  = lane & 15;
    const int kh   = lane >> 4;

    long long row = (long long)blockIdx.x * 128 + wv * 16 + m16;
    long long rclamp = (row < n_nodes) ? row : 0;  // clamp; invalid rows never stored
    float dg = deg[rclamp];
    float scale = 1.0f / fmaxf(dg, 1.0f);
    const float* pm = sum + rclamp * KD + kh * 8;
    const float* px = xin + rclamp * KD + kh * 8;

    v8f acc[NT];
#pragma unroll
    for (int ct = 0; ct < NT; ++ct) {
#pragma unroll
        for (int r = 0; r < 8; ++r) acc[ct][r] = 0.0f;
    }

    // wait for this wave's async copies, then barrier so all waves' LDS
    // writes are visible before fragment reads.
    asm volatile("s_wait_asynccnt 0" ::: "memory");
    __syncthreads();

#pragma unroll
    for (int ks = 0; ks < KD / 32; ++ks) {
        AF32 rm = load_a_raw(pm + ks * 32);   // mean rows (raw fp32)
        AF32 rx = load_a_raw(px + ks * 32);   // root rows (raw fp32)
        v16bf am = cvt_a_frag(rm, scale);
        v16bf ax = cvt_a_frag(rx, 1.0f);
        const __bf16* bl_base = sWl + (ks * 32 + kh * 8) * OUTC + m16;
        const __bf16* br_base = sWr + (ks * 32 + kh * 8) * OUTC + m16;
#pragma unroll
        for (int ct = 0; ct < NT; ++ct) {
            v16bf bl = load_b_frag(bl_base + ct * 16, OUTC);
            v16bf br = load_b_frag(br_base + ct * 16, OUTC);
            acc[ct] = __builtin_amdgcn_wmma_f32_16x16x32_bf16(
                false, am, false, bl, (short)0, acc[ct], false, false);
            acc[ct] = __builtin_amdgcn_wmma_f32_16x16x32_bf16(
                false, ax, false, br, (short)0, acc[ct], false, false);
        }
    }

    // C/D layout: lane<16 -> N=lane, M=r ; lane>=16 -> N=lane-16, M=8+r
    const int ncol = lane & 15;
#pragma unroll
    for (int ct = 0; ct < NT; ++ct) {
        int n = ct * 16 + ncol;
        float bv = bias[n];
#pragma unroll
        for (int r = 0; r < 8; ++r) {
            long long mrow = (long long)blockIdx.x * 128 + wv * 16 + kh * 8 + r;
            if (mrow < n_nodes) {
                float v = acc[ct][r] + bv;
                if (RELU) v = fmaxf(v, 0.0f);
                out[mrow * OUTC + n] = v;
            }
        }
    }
}

// ---------------------------------------------------------------------------
// launch
// ---------------------------------------------------------------------------
extern "C" void kernel_launch(void* const* d_in, const int* in_sizes, int n_in,
                              void* d_out, int out_size, void* d_ws, size_t ws_size,
                              hipStream_t stream) {
    const float*     x   = (const float*)d_in[0];
    const long long* ei  = (const long long*)d_in[1];
    const float*     W1l = (const float*)d_in[2];
    const float*     W1r = (const float*)d_in[3];
    const float*     b1  = (const float*)d_in[4];
    const float*     W2l = (const float*)d_in[5];
    const float*     W2r = (const float*)d_in[6];
    const float*     b2  = (const float*)d_in[7];
    float* out = (float*)d_out;

    const long long N = (long long)in_sizes[0] / IN_C;
    const long long E = (long long)in_sizes[1] / 2;

    float* sum = (float*)d_ws;          // N*128 f32
    float* deg = sum + N * IN_C;        // N f32
    float* h   = deg + N;               // N*128 f32
    __bf16* w1l = (__bf16*)(h + N * IN_C);      // 128*128 bf16
    __bf16* w1r = w1l + 128 * 128;
    __bf16* w2l = w1r + 128 * 128;              // 128*64 bf16
    __bf16* w2r = w2l + 128 * 64;

    const long long sumdeg = N * IN_C + N;
    int zgrid = (int)((sumdeg + 1023) / 1024);
    int sgrid = (int)((E * 32 + 255) / 256);
    int ggrid = (int)((N + 127) / 128);

    // one-time weight conversion fp32 -> bf16
    sage_w2bf<<<(128 * 128 + 255) / 256, 256, 0, stream>>>(W1l, w1l, 128 * 128);
    sage_w2bf<<<(128 * 128 + 255) / 256, 256, 0, stream>>>(W1r, w1r, 128 * 128);
    sage_w2bf<<<(128 * 64 + 255) / 256, 256, 0, stream>>>(W2l, w2l, 128 * 64);
    sage_w2bf<<<(128 * 64 + 255) / 256, 256, 0, stream>>>(W2r, w2r, 128 * 64);

    // ---- layer 1: mean-aggregate x, then h = relu(mean@W1l + x@W1r + b1)
    sage_zero_f32<<<zgrid, 256, 0, stream>>>(sum, sumdeg);
    sage_scatter<<<sgrid, 256, 0, stream>>>(x, ei, sum, deg, E);
    sage_gemm<HID_C, true><<<ggrid, 256, 0, stream>>>(sum, deg, x, w1l, w1r, b1, h, N);

    // ---- layer 2: mean-aggregate h, then out = mean@W2l + h@W2r + b2
    sage_zero_f32<<<zgrid, 256, 0, stream>>>(sum, sumdeg);
    sage_scatter<<<sgrid, 256, 0, stream>>>(h, ei, sum, deg, E);
    sage_gemm<64, false><<<ggrid, 256, 0, stream>>>(sum, deg, h, w2l, w2r, b2, out, N);
}
